// GCN_Net_86681029967974
// MI455X (gfx1250) — matrix operand order repaired and checked by reading, not compiled, not run on hardware
//
#include <hip/hip_runtime.h>
#include <hip/hip_bf16.h>

// ---------------------------------------------------------------------------
// GCN graph-matching net for MI455X (gfx1250, wave32, WMMA).
//
// Dominant op: x2 = (A_rownorm * K) @ x1  with K 4096x4096 f32, x1 4096x16.
// Row normalization is factored out:  diag(inv)·(K @ x1).  K tiles are
// converted f32->f16 in registers and fed to v_wmma_f32_16x16x32_f16 with
// f32 accumulation (memory-bound: ~8 FLOP/byte; K fits in the 192MB L2 so
// layers 2-3 re-read from L2, only layer 1 touches HBM).
//
// Split-K x8: 2048 waves (vs 256) so the memory-bound sweep has enough
// loads in flight to saturate HBM/L2 bandwidth; partials (2 MB) reduced by
// a tiny epilogue kernel that also applies inv_row and adds the ns term.
// ---------------------------------------------------------------------------

typedef __attribute__((ext_vector_type(16))) _Float16 v16h;
typedef __attribute__((ext_vector_type(8)))  float    v8f;
typedef __attribute__((ext_vector_type(4)))  float    v4f;

#define NN   4096      // N1*N2
#define CH   16        // gnn channel width
#define TAU_INV 20.0f  // 1/0.05
#define KSPLIT 8       // split-K factor
#define NTILE  (NN / 16)          // 256 output row-tiles
#define NWAVE  (NTILE * KSPLIT)   // 2048 GEMM waves

// --------------------------------------------------------------------------
// Kernel 0: per-row nonzero count of K -> inv_row[i] = 1/(cnt+1e-8),
// and extract diag(K) into column 0 of the node-feature buffer (stride 17).
// --------------------------------------------------------------------------
__global__ __launch_bounds__(256)
void k_rowstat(const float* __restrict__ K, float* __restrict__ inv_row,
               float* __restrict__ xbuf) {
    __shared__ int cnt[256];
    const int row = blockIdx.x;
    const float* kr = K + (long long)row * NN;
    int c = 0;
    for (int j = threadIdx.x; j < NN; j += 256)
        c += (kr[j] != 0.0f) ? 1 : 0;
    cnt[threadIdx.x] = c;
    __syncthreads();
    for (int s = 128; s > 0; s >>= 1) {
        if (threadIdx.x < s) cnt[threadIdx.x] += cnt[threadIdx.x + s];
        __syncthreads();
    }
    if (threadIdx.x == 0) {
        inv_row[row] = 1.0f / ((float)cnt[0] + 1e-8f);
        xbuf[row * 17] = kr[row];               // diag(K), initial x (N,1)
    }
}

// --------------------------------------------------------------------------
// Kernel 1: per-node 2-layer MLPs (nf and ns paths), 16 hidden units.
//   x1 = relu(relu(x@w1+b1)@w2+b2)  -> packed f16 in WMMA B-fragment order
//   xs = same for ns weights        -> f32 row-major (added after GEMM)
// B-fragment layout (16-bit, K=32): lanes 0-15 hold K=0..7,16..23 (lane=N, half
// idx=(K&7)+8*(K>>4)), lanes 16-31 hold K=8..15,24..31 (lane=16+N, half idx same).
// --------------------------------------------------------------------------
__global__ __launch_bounds__(256)
void k_mlp_pack(const float* __restrict__ xbuf, int in_n,
                const float* __restrict__ nf_w1, const float* __restrict__ nf_b1,
                const float* __restrict__ nf_w2, const float* __restrict__ nf_b2,
                const float* __restrict__ ns_w1, const float* __restrict__ ns_b1,
                const float* __restrict__ ns_w2, const float* __restrict__ ns_b2,
                _Float16* __restrict__ x1pack, float* __restrict__ xs) {
    const int i = blockIdx.x * 256 + threadIdx.x;      // node index
    float xin[17];
    for (int c = 0; c < in_n; ++c) xin[c] = xbuf[i * 17 + c];

    float h[CH], o[CH];
    // ---- nf path ----
    for (int n = 0; n < CH; ++n) {
        float a = nf_b1[n];
        for (int c = 0; c < in_n; ++c) a += xin[c] * nf_w1[c * CH + n];
        h[n] = fmaxf(a, 0.0f);
    }
    for (int n = 0; n < CH; ++n) {
        float a = nf_b2[n];
        for (int c = 0; c < CH; ++c) a += h[c] * nf_w2[c * CH + n];
        o[n] = fmaxf(a, 0.0f);
    }
    // scatter x1 row into B-fragment pack for its K-tile
    const int t  = i >> 5;          // K-tile (32 rows of x1 per tile)
    const int kk = i & 31;          // K index within tile
    const int laneBase = (kk & 8) ? 16 : 0;
    const int idx      = (kk & 7) | ((kk >> 1) & 8);
    _Float16* pb = x1pack + t * 512 + idx;
    for (int n = 0; n < CH; ++n)
        pb[(laneBase + n) * 16] = (_Float16)o[n];

    // ---- ns path ----
    for (int n = 0; n < CH; ++n) {
        float a = ns_b1[n];
        for (int c = 0; c < in_n; ++c) a += xin[c] * ns_w1[c * CH + n];
        h[n] = fmaxf(a, 0.0f);
    }
    for (int n = 0; n < CH; ++n) {
        float a = ns_b2[n];
        for (int c = 0; c < CH; ++c) a += h[c] * ns_w2[c * CH + n];
        xs[i * CH + n] = fmaxf(a, 0.0f);
    }
}

// --------------------------------------------------------------------------
// Kernel 2: WMMA GEMM, split-K.  One wave per (tile, ksplit): 2048 waves.
// A fragment: 16 rows of K, 32 K-values, converted f32->f16 in registers.
//   lanes 0-15: M=lane,    K = kt+{0..7, 16..23}
//   lanes16-31: M=lane-16, K = kt+{8..15, 24..31}
// Each wave stores its 16x16 f32 partial contiguously (32 B/lane).
// --------------------------------------------------------------------------
__global__ __launch_bounds__(128)
void k_gemm(const float* __restrict__ K,
            const _Float16* __restrict__ x1pack,
            float* __restrict__ part) {
    const int wave = threadIdx.x >> 5;
    const int lane = threadIdx.x & 31;
    const int w    = blockIdx.x * 4 + wave;        // 0..2047
    const int rt   = w & (NTILE - 1);              // output row-tile
    const int split = w >> 8;                      // K split 0..7
    const int m    = lane & 15;
    const int hs   = lane >> 4;                    // half-select 0/1
    const long long row = (long long)rt * 16 + m;  // A row for this lane
    const float* krow = K + row * NN + hs * 8;
    const _Float16* bp = x1pack + lane * 16;
    const int kt0 = split * (NN / 32 / KSPLIT);    // 16 K-tiles per split

    v8f c = {};
#pragma unroll 4
    for (int kk = 0; kk < NN / 32 / KSPLIT; ++kk) {
        const int kt = kt0 + kk;
        const float* p = krow + kt * 32;
        v4f f0 = *(const v4f*)(p);
        v4f f1 = *(const v4f*)(p + 4);
        v4f f2 = *(const v4f*)(p + 16);
        v4f f3 = *(const v4f*)(p + 20);
        v16h a;
#pragma unroll
        for (int e = 0; e < 4; ++e) {
            a[e]      = (_Float16)f0[e];
            a[4 + e]  = (_Float16)f1[e];
            a[8 + e]  = (_Float16)f2[e];
            a[12 + e] = (_Float16)f3[e];
        }
        v16h b = *(const v16h*)(bp + kt * 512);    // 32B aligned fragment
        c = __builtin_amdgcn_wmma_f32_16x16x32_f16(
                false, a, false, b, (short)0, c, false, false);
    }
    // contiguous per-lane partial store (2 x b128)
    *(v8f*)(part + ((size_t)w * 32 + lane) * 8) = c;
}

// --------------------------------------------------------------------------
// Kernel 2b: split-K reduction epilogue.
// x2[i][n] = inv_row[i] * sum_split part + xs[i][n]
// Fragment inverse map: i = rt*16 + v + 8*hs, n = lane&15.
// --------------------------------------------------------------------------
__global__ __launch_bounds__(256)
void k_reduce(const float* __restrict__ part, const float* __restrict__ inv_row,
              const float* __restrict__ xs, float* __restrict__ x2) {
    const int tid = blockIdx.x * 256 + threadIdx.x;   // 0..65535
    const int i = tid >> 4;                           // node row
    const int n = tid & 15;                           // channel
    const int rt = i >> 4;
    const int r  = i & 15;
    const int v  = r & 7;
    const int hs = r >> 3;
    const int lane = hs * 16 + n;
    float s = 0.0f;
#pragma unroll
    for (int sp = 0; sp < KSPLIT; ++sp)
        s += part[((size_t)(sp * NTILE + rt) * 32 + lane) * 8 + v];
    x2[i * CH + n] = inv_row[i] * s + xs[i * CH + n];
}

// --------------------------------------------------------------------------
// Kernel 3: classifier head  x3[i] = x2[i,:] @ cls_w + cls_b  (16 -> 1)
// --------------------------------------------------------------------------
__global__ __launch_bounds__(256)
void k_dot16(const float* __restrict__ x2, const float* __restrict__ w,
             const float* __restrict__ b, float* __restrict__ out) {
    const int i = blockIdx.x * 256 + threadIdx.x;
    float a = b[0];
    for (int n = 0; n < CH; ++n) a += x2[i * CH + n] * w[n];
    out[i] = a;
}

// --------------------------------------------------------------------------
// Kernel 4: log-space Sinkhorn, 64x64, 20 alternating row/col iters.
// One block of 64 threads; matrix lives in LDS (16 KB). transpose=1 reads
// src as v.reshape(64,64).T (final head).  dst = exp(log_s).
// --------------------------------------------------------------------------
__global__ __launch_bounds__(64)
void k_sinkhorn(const float* __restrict__ src, float* __restrict__ dst,
                int transpose) {
    __shared__ float ls[64 * 64];
    const int t = threadIdx.x;
    for (int idx = t; idx < 4096; idx += 64) {
        const int i = idx >> 6, j = idx & 63;
        float v = transpose ? src[j * 64 + i] : src[idx];
        ls[idx] = v * TAU_INV;
    }
    __syncthreads();
    for (int it = 0; it < 20; ++it) {
        if ((it & 1) == 0) {                // axis=1: row normalization
            float mx = -3.4e38f;
            for (int j = 0; j < 64; ++j) mx = fmaxf(mx, ls[t * 64 + j]);
            float s = 0.0f;
            for (int j = 0; j < 64; ++j) s += expf(ls[t * 64 + j] - mx);
            const float lse = mx + logf(s);
            for (int j = 0; j < 64; ++j) ls[t * 64 + j] -= lse;
        } else {                            // axis=0: column normalization
            float mx = -3.4e38f;
            for (int j = 0; j < 64; ++j) mx = fmaxf(mx, ls[j * 64 + t]);
            float s = 0.0f;
            for (int j = 0; j < 64; ++j) s += expf(ls[j * 64 + t] - mx);
            const float lse = mx + logf(s);
            for (int j = 0; j < 64; ++j) ls[j * 64 + t] -= lse;
        }
        __syncthreads();
    }
    for (int idx = t; idx < 4096; idx += 64)
        dst[idx] = expf(ls[idx]);
}

// --------------------------------------------------------------------------
// Kernel 5: concat  xbuf[i] = [x2[i,0:16], x4[i]]  (stride 17)
// --------------------------------------------------------------------------
__global__ __launch_bounds__(256)
void k_concat(const float* __restrict__ x2, const float* __restrict__ x4,
              float* __restrict__ xbuf) {
    const int i = blockIdx.x * 256 + threadIdx.x;
    for (int n = 0; n < CH; ++n) xbuf[i * 17 + n] = x2[i * CH + n];
    xbuf[i * 17 + CH] = x4[i];
}

// --------------------------------------------------------------------------
// Kernel 6: final FC  v[i] = xbuf[i,:17] @ fc_w + fc_b
// --------------------------------------------------------------------------
__global__ __launch_bounds__(256)
void k_fc(const float* __restrict__ xbuf, const float* __restrict__ w,
          const float* __restrict__ b, float* __restrict__ out) {
    const int i = blockIdx.x * 256 + threadIdx.x;
    float a = b[0];
    for (int c = 0; c < 17; ++c) a += xbuf[i * 17 + c] * w[c];
    out[i] = a;
}

// --------------------------------------------------------------------------
// Host side.
// Input order assumes JAX pytree flattening of setup_inputs() (dict keys
// sorted):  0:K  1:fc_b  2:fc_w  3+10l+{0:cls_b 1:cls_w 2:nf_b1 3:nf_b2
// 4:nf_w1 5:nf_w2 6:ns_b1 7:ns_b2 8:ns_w1 9:ns_w2}  33:n1  34:n2
// --------------------------------------------------------------------------
extern "C" void kernel_launch(void* const* d_in, const int* in_sizes, int n_in,
                              void* d_out, int out_size, void* d_ws, size_t ws_size,
                              hipStream_t stream) {
    (void)in_sizes; (void)n_in; (void)out_size; (void)ws_size;

    const float* Kp   = (const float*)d_in[0];
    const float* fc_b = (const float*)d_in[1];
    const float* fc_w = (const float*)d_in[2];

    // workspace carve-out (~3.1 MB total)
    float* inv_row = (float*)d_ws;                  // 4096
    float* xbuf    = inv_row + NN;                  // 4096*17
    float* xs      = xbuf + NN * 17;                // 4096*16
    float* x2      = xs + NN * CH;                  // 4096*16
    float* x3v     = x2 + NN * CH;                  // 4096
    float* x4      = x3v + NN;                      // 4096
    _Float16* x1pack = (_Float16*)(x4 + NN);        // 4096*16 f16 (32B aligned)
    float* part    = (float*)(x1pack + NN * CH);    // 2048*256 f32 partials

    k_rowstat<<<NN, 256, 0, stream>>>(Kp, inv_row, xbuf);

    for (int l = 0; l < 3; ++l) {
        const float* cls_b = (const float*)d_in[3 + l * 10 + 0];
        const float* cls_w = (const float*)d_in[3 + l * 10 + 1];
        const float* nf_b1 = (const float*)d_in[3 + l * 10 + 2];
        const float* nf_b2 = (const float*)d_in[3 + l * 10 + 3];
        const float* nf_w1 = (const float*)d_in[3 + l * 10 + 4];
        const float* nf_w2 = (const float*)d_in[3 + l * 10 + 5];
        const float* ns_b1 = (const float*)d_in[3 + l * 10 + 6];
        const float* ns_b2 = (const float*)d_in[3 + l * 10 + 7];
        const float* ns_w1 = (const float*)d_in[3 + l * 10 + 8];
        const float* ns_w2 = (const float*)d_in[3 + l * 10 + 9];
        const int in_n = (l == 0) ? 1 : (CH + 1);

        k_mlp_pack<<<NN / 256, 256, 0, stream>>>(xbuf, in_n,
            nf_w1, nf_b1, nf_w2, nf_b2, ns_w1, ns_b1, ns_w2, ns_b2,
            x1pack, xs);
        k_gemm<<<NWAVE / 4, 128, 0, stream>>>(Kp, x1pack, part);
        k_reduce<<<NN * CH / 256, 256, 0, stream>>>(part, inv_row, xs, x2);
        k_dot16<<<NN / 256, 256, 0, stream>>>(x2, cls_w, cls_b, x3v);
        k_sinkhorn<<<1, 64, 0, stream>>>(x3v, x4, 0);
        k_concat<<<NN / 256, 256, 0, stream>>>(x2, x4, xbuf);
    }

    k_fc<<<NN / 256, 256, 0, stream>>>(xbuf, fc_w, fc_b, x3v);
    k_sinkhorn<<<1, 64, 0, stream>>>(x3v, (float*)d_out, 1);
}
